// CrossAttentionLayer_9131100471663
// MI455X (gfx1250) — matrix operand order repaired and checked by reading, not compiled, not run on hardware
//
#include <hip/hip_runtime.h>
#include <hip/hip_bf16.h>
#include <cstdint>
#include <cstddef>

typedef _Float16 h16;
typedef __attribute__((ext_vector_type(16))) _Float16 v16h;
typedef __attribute__((ext_vector_type(8)))  _Float16 v8h;
typedef __attribute__((ext_vector_type(8)))  float    v8f;

#define DEVINL __device__ __forceinline__

constexpr int CB   = 16;          // batch
constexpr int CS   = 512;         // sequence length
constexpr int CH   = 512;         // hidden
constexpr int CNH  = 8;           // heads
constexpr int CDH  = 64;          // head dim
constexpr int CMEM = 16;          // memory tokens
constexpr int CMM  = CS + CMEM;   // 528
constexpr int CMMP = 544;         // padded K (multiple of 32) for apply GEMMs
constexpr int CH3  = 3 * CH;      // 1536
constexpr float CNEG = -1e9f;

// ---------------------------------------------------------------------------
// WMMA fragment helpers (wave32, V_WMMA_F32_16X16X32_F16 layouts per ISA 7.12.2)
//  A (16x32 f16): lane holds row M = lane&15; halves e=0..7 -> K = 8*(lane>>4)+e,
//                 e=8..15 -> K = 16 + 8*(lane>>4) + (e-8)   (two contiguous 8-half runs)
//  B (32x16 f16): lane holds col N = lane&15; same K order as A
//  C/D (16x16 f32): vgpr v -> M = v + 8*(lane>>4), N = lane&15
// ---------------------------------------------------------------------------
DEVINL v16h frag_a(const h16* __restrict__ row, int k0, int lane) {
  int hk = (lane >> 4) << 3;                       // 0 or 8
  v8h r0 = *(const v8h*)(row + k0 + hk);
  v8h r1 = *(const v8h*)(row + k0 + hk + 16);
  v16h a;
#pragma unroll
  for (int e = 0; e < 8; ++e) { a[e] = r0[e]; a[e + 8] = r1[e]; }
  return a;
}

DEVINL v16h frag_b(const h16* __restrict__ krow, int k0, int lane) {
  int kb = k0 + ((lane >> 4) << 3);                // +0 or +8
  v8h s0 = *(const v8h*)(krow + kb);
  v8h s1 = *(const v8h*)(krow + kb + 16);
  v16h b;
#pragma unroll
  for (int e = 0; e < 8; ++e) { b[e] = s0[e]; b[e + 8] = s1[e]; }
  return b;
}

DEVINL v8f wmma_step(v16h a, v16h b, v8f c) {
  return __builtin_amdgcn_wmma_f32_16x16x32_f16(false, a, false, b, (short)0, c,
                                                false, false);
}

// ---------------------------------------------------------------------------
// f32 -> f16 conversion
// ---------------------------------------------------------------------------
__global__ void k_cvt_f16(const float* __restrict__ src, h16* __restrict__ dst, int n) {
  int i = blockIdx.x * blockDim.x + threadIdx.x;
  if (i < n) dst[i] = (h16)src[i];
}

// Build transposed concat:  dstT[b][c][m] = (m<16 ? mem[m][c] : (m<528 ? x[b][m-16][c] : 0))
__global__ void k_catT(const float* __restrict__ mem, const float* __restrict__ x,
                       h16* __restrict__ dstT) {
  int i = blockIdx.x * blockDim.x + threadIdx.x;   // over CB*CH*CMMP
  if (i >= CB * CH * CMMP) return;
  int m   = i % CMMP;
  int rem = i / CMMP;
  int c   = rem % CH;
  int b   = rem / CH;
  float val = 0.f;
  if (m < CMEM)     val = mem[m * CH + c];
  else if (m < CMM) val = x[((size_t)b * CS + (m - CMEM)) * CH + c];
  dstT[i] = (h16)val;
}

// ---------------------------------------------------------------------------
// Projection: outP[b][m][o] = sum_i concat(mem,x)[b][m][i] * W[o][i]   (f16 out)
// One wave computes a 16x64 output tile (A-fragment reused across 4 N-tiles).
// ---------------------------------------------------------------------------
__global__ void k_proj(const h16* __restrict__ memH, const h16* __restrict__ xH,
                       const h16* __restrict__ wH, h16* __restrict__ outP) {
  const int MT = CMM / 16, NG = (CH / 16) / 4;     // 33, 8
  int wv = blockIdx.x * (blockDim.x >> 5) + (threadIdx.x >> 5);
  if (wv >= CB * MT * NG) return;
  int ng = wv % NG; int t = wv / NG;
  int mt = t % MT;  int b = t / MT;
  int lane = threadIdx.x & 31;
  int mr = mt * 16 + (lane & 15);
  const h16* rowA = (mr < CMEM) ? (memH + (size_t)mr * CH)
                                : (xH + ((size_t)b * CS + (mr - CMEM)) * CH);
  const h16* rowB[4];
#pragma unroll
  for (int j = 0; j < 4; ++j)
    rowB[j] = wH + (size_t)(ng * 64 + j * 16 + (lane & 15)) * CH;
  v8f acc[4] = {};
#pragma unroll 2
  for (int k0 = 0; k0 < CH; k0 += 32) {
    v16h a = frag_a(rowA, k0, lane);
#pragma unroll
    for (int j = 0; j < 4; ++j)
      acc[j] = wmma_step(a, frag_b(rowB[j], k0, lane), acc[j]);
  }
  int mbase = mt * 16 + ((lane >> 4) << 3);
#pragma unroll
  for (int j = 0; j < 4; ++j) {
    int n = ng * 64 + j * 16 + (lane & 15);
#pragma unroll
    for (int r = 0; r < 8; ++r)
      outP[((size_t)b * CMM + (mbase + r)) * CH + n] = (h16)acc[j][r];
  }
}

// ---------------------------------------------------------------------------
// Affinity: aff[b][h][m][n] = Xp[b][m][h*64+d] . Yp[b][n][h*64+d], masked.
// One wave computes a 16x48 tile (A-fragment reused across 3 N-tiles; 33 % 3 == 0).
// ---------------------------------------------------------------------------
__global__ void k_aff(const h16* __restrict__ Xp, const h16* __restrict__ Yp,
                      const int* __restrict__ mx, const int* __restrict__ my,
                      float* __restrict__ aff) {
  const int T = CMM / 16, NG = T / 3;              // 33, 11
  int wv = blockIdx.x * (blockDim.x >> 5) + (threadIdx.x >> 5);
  if (wv >= CB * CNH * T * NG) return;
  int ng = wv % NG; int t = wv / NG;
  int mt = t % T;   t /= T;
  int h  = t % CNH; int b = t / CNH;
  int lane = threadIdx.x & 31;
  const h16* rowA = Xp + ((size_t)b * CMM + (mt * 16 + (lane & 15))) * CH + h * CDH;
  const h16* rowB[3];
#pragma unroll
  for (int j = 0; j < 3; ++j)
    rowB[j] = Yp + ((size_t)b * CMM + ((ng * 3 + j) * 16 + (lane & 15))) * CH + h * CDH;
  v8f acc[3] = {};
#pragma unroll
  for (int k0 = 0; k0 < CDH; k0 += 32) {
    v16h a = frag_a(rowA, k0, lane);
#pragma unroll
    for (int j = 0; j < 3; ++j)
      acc[j] = wmma_step(a, frag_b(rowB[j], k0, lane), acc[j]);
  }
  int mbase = mt * 16 + ((lane >> 4) << 3);
  float* out = aff + ((size_t)(b * CNH + h) * CMM) * CMM;
#pragma unroll
  for (int j = 0; j < 3; ++j) {
    int n = (ng * 3 + j) * 16 + (lane & 15);
    bool nok = (n < CMEM) || (my[b * CS + (n - CMEM)] != 0);
#pragma unroll
    for (int r = 0; r < 8; ++r) {
      int m = mbase + r;
      bool mok = (m < CMEM) || (mx[b * CS + (m - CMEM)] != 0);
      out[(size_t)m * CMM + n] = (mok && nok) ? acc[j][r] : CNEG;
    }
  }
}

// ---------------------------------------------------------------------------
// Softmax over n (axis=3), head mean -> probs[b][m][n] (f16, K padded to 544)
// block = (b,m), 8 waves = 8 heads
// ---------------------------------------------------------------------------
__global__ void k_smax_rows(const float* __restrict__ aff, h16* __restrict__ outA) {
  __shared__ float acc[CMMP];
  int b = blockIdx.x / CMM, m = blockIdx.x % CMM;
  for (int i = threadIdx.x; i < CMMP; i += blockDim.x) acc[i] = 0.f;
  __syncthreads();
  int h = threadIdx.x >> 5, lane = threadIdx.x & 31;
  const float* row = aff + ((size_t)(b * CNH + h) * CMM + m) * CMM;
  float mx = -3.0e38f;
  for (int n = lane; n < CMM; n += 32) mx = fmaxf(mx, row[n]);
#pragma unroll
  for (int o = 16; o >= 1; o >>= 1) mx = fmaxf(mx, __shfl_xor(mx, o, 32));
  float sm = 0.f;
  for (int n = lane; n < CMM; n += 32) sm += __expf(row[n] - mx);
#pragma unroll
  for (int o = 16; o >= 1; o >>= 1) sm += __shfl_xor(sm, o, 32);
  float inv = 0.125f / sm;                         // mean over 8 heads folded in
  for (int n = lane; n < CMM; n += 32) atomicAdd(&acc[n], __expf(row[n] - mx) * inv);
  __syncthreads();
  h16* out = outA + ((size_t)b * CMM + m) * CMMP;
  for (int i = threadIdx.x; i < CMMP; i += blockDim.x) out[i] = (h16)acc[i];
}

// Softmax over m (axis=2), head mean -> probsT[b][n][m] (transposed store)
__global__ void k_smax_cols(const float* __restrict__ aff, h16* __restrict__ outA) {
  __shared__ float acc[CMMP];
  int b = blockIdx.x / CMM, n = blockIdx.x % CMM;
  for (int i = threadIdx.x; i < CMMP; i += blockDim.x) acc[i] = 0.f;
  __syncthreads();
  int h = threadIdx.x >> 5, lane = threadIdx.x & 31;
  const float* col = aff + ((size_t)(b * CNH + h) * CMM) * CMM + n;
  float mx = -3.0e38f;
  for (int m = lane; m < CMM; m += 32) mx = fmaxf(mx, col[(size_t)m * CMM]);
#pragma unroll
  for (int o = 16; o >= 1; o >>= 1) mx = fmaxf(mx, __shfl_xor(mx, o, 32));
  float sm = 0.f;
  for (int m = lane; m < CMM; m += 32) sm += __expf(col[(size_t)m * CMM] - mx);
#pragma unroll
  for (int o = 16; o >= 1; o >>= 1) sm += __shfl_xor(sm, o, 32);
  float inv = 0.125f / sm;
  for (int m = lane; m < CMM; m += 32)
    atomicAdd(&acc[m], __expf(col[(size_t)m * CMM] - mx) * inv);
  __syncthreads();
  h16* out = outA + ((size_t)b * CMM + n) * CMMP;
  for (int i = threadIdx.x; i < CMMP; i += blockDim.x) out[i] = (h16)acc[i];
}

// ---------------------------------------------------------------------------
// Apply: out[b][r-16][c] = sum_k probs[b][r][k] * catT[b][c][k], f16 out.
// One wave computes a 16x64 output tile (A reused across 4 N-tiles).
// ---------------------------------------------------------------------------
__global__ void k_apply(const h16* __restrict__ probs,   // [CB][CMM][CMMP]
                        const h16* __restrict__ catT,    // [CB][CH][CMMP]
                        h16* __restrict__ out) {         // [CB][CS][CH]
  const int RT = CMM / 16, NG = (CH / 16) / 4;     // 33, 8
  int wv = blockIdx.x * (blockDim.x >> 5) + (threadIdx.x >> 5);
  if (wv >= CB * RT * NG) return;
  int ng = wv % NG; int t = wv / NG;
  int rt = t % RT;  int b = t / RT;
  int lane = threadIdx.x & 31;
  const h16* rowA = probs + ((size_t)b * CMM + (rt * 16 + (lane & 15))) * CMMP;
  const h16* rowB[4];
#pragma unroll
  for (int j = 0; j < 4; ++j)
    rowB[j] = catT + ((size_t)b * CH + (ng * 64 + j * 16 + (lane & 15))) * CMMP;
  v8f acc[4] = {};
  for (int k0 = 0; k0 < CMMP; k0 += 32) {
    v16h a = frag_a(rowA, k0, lane);
#pragma unroll
    for (int j = 0; j < 4; ++j)
      acc[j] = wmma_step(a, frag_b(rowB[j], k0, lane), acc[j]);
  }
  int rbase = rt * 16 + ((lane >> 4) << 3);
#pragma unroll
  for (int j = 0; j < 4; ++j) {
    int c = ng * 64 + j * 16 + (lane & 15);
#pragma unroll
    for (int r = 0; r < 8; ++r) {
      int rr = rbase + r;
      if (rr >= CMEM)
        out[((size_t)b * CS + (rr - CMEM)) * CH + c] = (h16)acc[j][r];
    }
  }
}

// ---------------------------------------------------------------------------
// MLP: act[b][s][o] = relu( concat(s0,s1,s2)[b][s][:] . W[o][:] + bias[o] )
// One wave computes a 16x64 output tile.
// ---------------------------------------------------------------------------
__global__ void k_mlp(const h16* __restrict__ s0, const h16* __restrict__ s1,
                      const h16* __restrict__ s2, const h16* __restrict__ wH,
                      const float* __restrict__ bias, float* __restrict__ act) {
  const int MT = CS / 16, NG = (CH / 16) / 4;      // 32, 8
  int wv = blockIdx.x * (blockDim.x >> 5) + (threadIdx.x >> 5);
  if (wv >= CB * MT * NG) return;
  int ng = wv % NG; int t = wv / NG;
  int mt = t % MT;  int b = t / MT;
  int lane = threadIdx.x & 31;
  size_t rowoff = ((size_t)b * CS + (mt * 16 + (lane & 15))) * CH;
  const h16* segs[3] = { s0 + rowoff, s1 + rowoff, s2 + rowoff };
  const h16* rowB[4];
#pragma unroll
  for (int j = 0; j < 4; ++j)
    rowB[j] = wH + (size_t)(ng * 64 + j * 16 + (lane & 15)) * CH3;
  int hk = (lane >> 4) << 3;
  v8f acc[4] = {};
  for (int k0 = 0; k0 < CH3; k0 += 32) {
    int ka = k0 + hk, kb = ka + 16;                // 8-half runs never straddle a segment
    v8h r0 = *(const v8h*)(segs[ka >> 9] + (ka & 511));
    v8h r1 = *(const v8h*)(segs[kb >> 9] + (kb & 511));
    v16h a;
#pragma unroll
    for (int e = 0; e < 8; ++e) { a[e] = r0[e]; a[e + 8] = r1[e]; }
#pragma unroll
    for (int j = 0; j < 4; ++j)
      acc[j] = wmma_step(a, frag_b(rowB[j], k0, lane), acc[j]);
  }
  int sbase = mt * 16 + ((lane >> 4) << 3);
#pragma unroll
  for (int j = 0; j < 4; ++j) {
    int o = ng * 64 + j * 16 + (lane & 15);
    float bv = bias[o];
#pragma unroll
    for (int r = 0; r < 8; ++r) {
      float v = acc[j][r] + bv;
      act[((size_t)b * CS + (sbase + r)) * CH + o] = v > 0.f ? v : 0.f;
    }
  }
}

// ---------------------------------------------------------------------------
// Residual + LayerNorm: out = LN(x + act) * g + be      (block = one row, 256 thr)
// ---------------------------------------------------------------------------
__global__ void k_ln(const float* __restrict__ x, const float* __restrict__ a,
                     const float* __restrict__ g, const float* __restrict__ be,
                     float* __restrict__ out) {
  __shared__ float red[256];
  int row = blockIdx.x;
  const float* xr = x + (size_t)row * CH;
  const float* ar = a + (size_t)row * CH;
  int t = threadIdx.x;
  float y0 = xr[t] + ar[t], y1 = xr[t + 256] + ar[t + 256];
  red[t] = y0 + y1; __syncthreads();
  for (int s = 128; s > 0; s >>= 1) { if (t < s) red[t] += red[t + s]; __syncthreads(); }
  float mu = red[0] * (1.0f / CH); __syncthreads();
  float d0 = y0 - mu, d1 = y1 - mu;
  red[t] = d0 * d0 + d1 * d1; __syncthreads();
  for (int s = 128; s > 0; s >>= 1) { if (t < s) red[t] += red[t + s]; __syncthreads(); }
  float inv = rsqrtf(red[0] * (1.0f / CH) + 1e-5f);
  float* o = out + (size_t)row * CH;
  o[t]       = d0 * inv * g[t]       + be[t];
  o[t + 256] = d1 * inv * g[t + 256] + be[t + 256];
}

// ---------------------------------------------------------------------------
extern "C" void kernel_launch(void* const* d_in, const int* in_sizes, int n_in,
                              void* d_out, int out_size, void* d_ws, size_t ws_size,
                              hipStream_t stream) {
  (void)in_sizes; (void)n_in; (void)out_size; (void)ws_size;
  const float* xin[3]  = { (const float*)d_in[0], (const float*)d_in[1], (const float*)d_in[2] };
  const int*   msk[3]  = { (const int*)d_in[3], (const int*)d_in[4], (const int*)d_in[5] };
  const float* axm[3]  = { (const float*)d_in[6],  (const float*)d_in[10], (const float*)d_in[14] };
  const float* aym[3]  = { (const float*)d_in[7],  (const float*)d_in[11], (const float*)d_in[15] };
  const float* awx[3]  = { (const float*)d_in[8],  (const float*)d_in[12], (const float*)d_in[16] };
  const float* awy[3]  = { (const float*)d_in[9],  (const float*)d_in[13], (const float*)d_in[17] };
  const float* wml[3]  = { (const float*)d_in[18], (const float*)d_in[22], (const float*)d_in[26] };
  const float* bml[3]  = { (const float*)d_in[19], (const float*)d_in[23], (const float*)d_in[27] };
  const float* gln[3]  = { (const float*)d_in[20], (const float*)d_in[24], (const float*)d_in[28] };
  const float* beln[3] = { (const float*)d_in[21], (const float*)d_in[25], (const float*)d_in[29] };

  size_t off = 0;
  auto alloc = [&](size_t bytes) -> void* {
    off = (off + 255) & ~(size_t)255;
    void* p = (char*)d_ws + off;
    off += bytes;
    return p;
  };
  const size_t NBSH = (size_t)CB * CS * CH;        // 4,194,304

  h16* xh[3];  for (int s = 0; s < 3; ++s) xh[s]  = (h16*)alloc(NBSH * 2);
  h16* mxh[3]; for (int p = 0; p < 3; ++p) mxh[p] = (h16*)alloc((size_t)CMEM * CH * 2);
  h16* myh[3]; for (int p = 0; p < 3; ++p) myh[p] = (h16*)alloc((size_t)CMEM * CH * 2);
  h16* wxh[3]; for (int p = 0; p < 3; ++p) wxh[p] = (h16*)alloc((size_t)CH * CH * 2);
  h16* wyh[3]; for (int p = 0; p < 3; ++p) wyh[p] = (h16*)alloc((size_t)CH * CH * 2);
  h16* wmh[3]; for (int s = 0; s < 3; ++s) wmh[s] = (h16*)alloc((size_t)CH * CH3 * 2);
  h16* Xp   = (h16*)alloc((size_t)CB * CMM * CH * 2);
  h16* Yp   = (h16*)alloc((size_t)CB * CMM * CH * 2);
  h16* XT   = (h16*)alloc((size_t)CB * CH * CMMP * 2);
  h16* YT   = (h16*)alloc((size_t)CB * CH * CMMP * 2);
  h16* pXYT = (h16*)alloc((size_t)CB * CMM * CMMP * 2);
  h16* pYX  = (h16*)alloc((size_t)CB * CMM * CMMP * 2);
  h16* P[3][2];
  for (int p = 0; p < 3; ++p) for (int s = 0; s < 2; ++s) P[p][s] = (h16*)alloc(NBSH * 2);
  float* aff = (float*)alloc((size_t)CB * CNH * CMM * CMM * 4);
  // act buffers reuse the aff region (MLP runs after all attention passes)
  float* act[3] = { aff, aff + NBSH, aff + 2 * NBSH };

  auto cvt = [&](const float* s, h16* d, int n) {
    k_cvt_f16<<<(n + 255) / 256, 256, 0, stream>>>(s, d, n);
  };
  for (int s = 0; s < 3; ++s) cvt(xin[s], xh[s], (int)NBSH);
  for (int p = 0; p < 3; ++p) {
    cvt(axm[p], mxh[p], CMEM * CH);
    cvt(aym[p], myh[p], CMEM * CH);
    cvt(awx[p], wxh[p], CH * CH);
    cvt(awy[p], wyh[p], CH * CH);
  }
  for (int s = 0; s < 3; ++s) cvt(wml[s], wmh[s], CH * CH3);

  // pair -> (x stream, y stream)
  const int PX[3] = { 0, 0, 1 };                   // im, im, qe
  const int PY[3] = { 1, 2, 2 };                   // qe, hi, hi

  const int projBlocks  = (CB * (CMM / 16) * ((CH / 16) / 4)) / 8;        // 528
  const int affBlocks   = (CB * CNH * (CMM / 16) * ((CMM / 16) / 3)) / 8; // 5808
  const int applyBlocks = (CB * (CMM / 16) * ((CH / 16) / 4)) / 8;        // 528
  const int mlpBlocks   = (CB * (CS / 16) * ((CH / 16) / 4)) / 8;         // 512
  const int catTn       = CB * CH * CMMP;

  for (int p = 0; p < 3; ++p) {
    const int xs = PX[p], ys = PY[p];
    k_proj<<<projBlocks, 256, 0, stream>>>(mxh[p], xh[xs], wxh[p], Xp);
    k_proj<<<projBlocks, 256, 0, stream>>>(myh[p], xh[ys], wyh[p], Yp);
    k_catT<<<(catTn + 255) / 256, 256, 0, stream>>>(axm[p], xin[xs], XT);
    k_catT<<<(catTn + 255) / 256, 256, 0, stream>>>(aym[p], xin[ys], YT);
    k_aff<<<affBlocks, 256, 0, stream>>>(Xp, Yp, msk[xs], msk[ys], aff);
    k_smax_rows<<<CB * CMM, 256, 0, stream>>>(aff, pYX);    // softmax over n -> aYX
    k_smax_cols<<<CB * CMM, 256, 0, stream>>>(aff, pXYT);   // softmax over m -> aXY^T
    k_apply<<<applyBlocks, 256, 0, stream>>>(pXYT, XT, P[p][0]);   // XinY
    k_apply<<<applyBlocks, 256, 0, stream>>>(pYX,  YT, P[p][1]);   // YinX
  }

  // MLP concat segments per stream:
  //  im: [im, qe_in_im(P0yx), hi_in_im(P1yx)]
  //  qe: [qe, hi_in_qe(P2yx), im_in_qe(P0xy)]
  //  hi: [hi, qe_in_hi(P2xy), im_in_hi(P1xy)]
  const h16* seg1[3] = { P[0][1], P[2][1], P[2][0] };
  const h16* seg2[3] = { P[1][1], P[0][0], P[1][0] };
  float* outp = (float*)d_out;
  for (int s = 0; s < 3; ++s) {
    k_mlp<<<mlpBlocks, 256, 0, stream>>>(xh[s], seg1[s], seg2[s], wmh[s], bml[s], act[s]);
    k_ln<<<CB * CS, 256, 0, stream>>>(xin[s], act[s], gln[s], beln[s], outp + s * NBSH);
  }
}
